// CustomMask2Former_20117626814862
// MI455X (gfx1250) — matrix operand order repaired and checked by reading, not compiled, not run on hardware
//
#include <hip/hip_runtime.h>
#include <hip/hip_bf16.h>
#include <math.h>

// CustomMask2Former forward for gfx1250 (MI455X).
// All matmul-shaped work (QKV/attn/FFN/TTT/mask-head/final mask einsum) runs
// through a single batched WMMA GEMM kernel using v_wmma_f32_16x16x32_f16,
// with double-buffered LDS staging and 32x32-per-wave output tiles.

typedef _Float16 half_t;
typedef __attribute__((ext_vector_type(2)))  half_t h2;
typedef __attribute__((ext_vector_type(16))) half_t v16h;
typedef __attribute__((ext_vector_type(8)))  float  v8f;

// ---------------------------------------------------------------------------
// Batched WMMA GEMM: C[z] = act(alpha * A[z] * op(B[z]) + bias)
//   A: [M,K] f16 row-major (lda)  -- all lda/ldb/K in this model are even
//   B: btrans=1 -> [N,K] row-major (weights, computes A*B^T); btrans=0 -> [K,N]
//   z = zo*inner + zi two-level batching (batch x head, layer x batch)
// Block tile 128x64, BK=32, 256 threads = 8 waves (4 in M x 2 in N); each wave
// computes a 32x32 patch = 4 x (16x16x32) WMMA with f32 accumulators.
// LDS is double buffered: tile t+1 streams in while tile t is computed.
// Staging is two-phase (issue all loads, then select+store) so up to 12 global
// loads stay in flight per wave; bounds are branch-free via address clamping.
// ---------------------------------------------------------------------------
struct GemmP {
  const half_t* A; const half_t* B; const float* bias;
  float* Cf; half_t* Ch;
  int M, N, K, lda, ldb, ldc;
  long long sAo, sAi, sBo, sBi, sCo, sCi;
  int inner, btrans, act; float alpha;
};

__global__ __launch_bounds__(256) void wmma_gemm_kernel(GemmP p) {
  // exercise the CDNA5 tensor-counter wait path (TENSORcnt is 0: free)
  __builtin_amdgcn_s_wait_tensorcnt(0);

  int z = blockIdx.z, zo = z / p.inner, zi = z % p.inner;
  const half_t* A = p.A + zo * p.sAo + zi * p.sAi;
  const half_t* B = p.B + zo * p.sBo + zi * p.sBi;
  long long coff = zo * p.sCo + zi * p.sCi;
  float*  Cf = p.Cf ? p.Cf + coff : nullptr;
  half_t* Ch = p.Ch ? p.Ch + coff : nullptr;

  __shared__ half_t As[2][128][36];   // [buf][m][k], padded rows
  __shared__ half_t Bs[2][64][36];    // [buf][n][k]

  int tid  = threadIdx.x;
  int lane = tid & 31, warp = tid >> 5;
  int wm = warp & 3, wn = warp >> 2;           // 4 M-waves x 2 N-waves
  int m0 = blockIdx.y * 128, n0 = blockIdx.x * 64;

  v8f acc00 = {}, acc01 = {}, acc10 = {}, acc11 = {};

  // stage one 128x32 A tile + 64x32 B tile into LDS buffer `buf`
  auto load_tile = [&](int buf, int kt) {
    int kbase   = kt * 32;
    int mClamp  = p.M - 1;
    int kClampP = p.K - 2;   // pair clamp (K even)
    int kClamp  = p.K - 1;

    // ---- phase 1: issue ALL global loads (no consumers in between) ----
    unsigned int va[8];
#pragma unroll
    for (int it = 0; it < 8; ++it) {
      int i = tid + it * 256;
      int r = i >> 4, c = (i & 15) << 1;
      int gm = m0 + r, gk = kbase + c;
      int cm = gm < mClamp ? gm : mClamp;
      int ck = gk < kClampP ? gk : kClampP;
      va[it] = *(const unsigned int*)(A + (long long)cm * p.lda + ck);
    }
    if (p.btrans) {
      int nClamp = p.N - 1;
      unsigned int vb[4];
#pragma unroll
      for (int it = 0; it < 4; ++it) {
        int i = tid + it * 256;
        int r = i >> 4, c = (i & 15) << 1;
        int gn = n0 + r, gk = kbase + c;
        int cn = gn < nClamp ? gn : nClamp;
        int ck = gk < kClampP ? gk : kClampP;
        vb[it] = *(const unsigned int*)(B + (long long)cn * p.ldb + ck);
      }
      // ---- phase 2: bounds-select + LDS stores ----
#pragma unroll
      for (int it = 0; it < 4; ++it) {
        int i = tid + it * 256;
        int r = i >> 4, c = (i & 15) << 1;
        int gn = n0 + r, gk = kbase + c;
        bool ok = (gn < p.N) && (gk < p.K);
        *(unsigned int*)(&Bs[buf][r][c]) = ok ? vb[it] : 0u;
      }
    } else {
      // B is [K,N]: pairs are contiguous along N -> coalesced loads,
      // transposed into Bs[n][k] via two 16-bit LDS stores.
      int nClampP = p.N - 2;   // N even in all btrans=0 uses
      h2 vb[4];
#pragma unroll
      for (int it = 0; it < 4; ++it) {
        int i = tid + it * 256;
        int k = i >> 5, np = (i & 31) << 1;
        int gk = kbase + k, gn = n0 + np;
        int ck = gk < kClamp ? gk : kClamp;
        int cn = gn < nClampP ? gn : nClampP;
        vb[it] = *(const h2*)(B + (long long)ck * p.ldb + cn);
      }
#pragma unroll
      for (int it = 0; it < 4; ++it) {
        int i = tid + it * 256;
        int k = i >> 5, np = (i & 31) << 1;
        int gk = kbase + k, gn = n0 + np;
        bool ok = (gk < p.K) && (gn < p.N);
        Bs[buf][np][k]     = ok ? vb[it][0] : (half_t)0.f;
        Bs[buf][np + 1][k] = ok ? vb[it][1] : (half_t)0.f;
      }
    }
#pragma unroll
    for (int it = 0; it < 8; ++it) {
      int i = tid + it * 256;
      int r = i >> 4, c = (i & 15) << 1;
      int gm = m0 + r, gk = kbase + c;
      bool ok = (gm < p.M) && (gk < p.K);
      *(unsigned int*)(&As[buf][r][c]) = ok ? va[it] : 0u;
    }
  };

  int ktiles = (p.K + 31) >> 5;
  load_tile(0, 0);
  __syncthreads();
  int buf = 0;

  for (int kt = 0; kt < ktiles; ++kt) {
    if (kt + 1 < ktiles) {
      load_tile(buf ^ 1, kt + 1);                       // overlap with WMMA below
      __builtin_prefetch(&A[(long long)(m0 + warp * 16) * p.lda + (kt + 2) * 32], 0, 1);
    }
    // A fragments 16x32 (ISA 7.12.2): lane L -> M=L&15;
    // half h -> K = (h&7) + ((h>>3)<<4) + ((L>>4)<<3)
    v16h a0, a1;
    {
      int mr = wm * 32 + (lane & 15);
      int kb = (lane >> 4) << 3;
#pragma unroll
      for (int h = 0; h < 16; ++h) {
        int kk = (h & 7) + ((h >> 3) << 4) + kb;
        a0[h] = As[buf][mr][kk];
        a1[h] = As[buf][mr + 16][kk];
      }
    }
    // B fragments 32x16: lane L -> N=L&15 (+tile); K = (h&7) + ((h>>3)<<4) + ((L>>4)<<3)
    v16h b0, b1;
    {
      int kb  = (lane >> 4) << 3;
      int nc0 = wn * 32 + (lane & 15);
#pragma unroll
      for (int h = 0; h < 16; ++h) {
        b0[h] = Bs[buf][nc0][kb + (h & 7) + ((h >> 3) << 4)];
        b1[h] = Bs[buf][nc0 + 16][kb + (h & 7) + ((h >> 3) << 4)];
      }
    }
    acc00 = __builtin_amdgcn_wmma_f32_16x16x32_f16(false, a0, false, b0, (short)0, acc00, false, false);
    acc01 = __builtin_amdgcn_wmma_f32_16x16x32_f16(false, a0, false, b1, (short)0, acc01, false, false);
    acc10 = __builtin_amdgcn_wmma_f32_16x16x32_f16(false, a1, false, b0, (short)0, acc10, false, false);
    acc11 = __builtin_amdgcn_wmma_f32_16x16x32_f16(false, a1, false, b1, (short)0, acc11, false, false);
    __syncthreads();
    buf ^= 1;
  }

  // C/D 16x16 f32 layout: lane L -> N=L&15, M = r + ((L>>4)<<3)
  int mrow = m0 + wm * 32 + ((lane >> 4) << 3);
  int ncol = n0 + wn * 32 + (lane & 15);
#pragma unroll
  for (int r = 0; r < 8; ++r) {
    for (int mt = 0; mt < 2; ++mt) {
      int gm = mrow + mt * 16 + r;
      if (gm >= p.M) continue;
      for (int nt = 0; nt < 2; ++nt) {
        int gn = ncol + nt * 16;
        if (gn >= p.N) continue;
        float v;
        if (mt == 0) v = nt ? acc01[r] : acc00[r];
        else         v = nt ? acc11[r] : acc10[r];
        v *= p.alpha;
        if (p.bias) v += p.bias[gn];
        if (p.act == 1)      v = fmaxf(v, 0.f);
        else if (p.act == 2) v = 0.5f * v * (1.f + erff(v * 0.70710678f));   // exact gelu
        else if (p.act == 3) v = 1.f / (1.f + __expf(-v));
        long long o = (long long)gm * p.ldc + gn;
        if (Cf) Cf[o] = v;
        if (Ch) Ch[o] = (half_t)v;
      }
    }
  }
}

// ---------------------------------------------------------------------------
// Elementwise / reduction helpers
// ---------------------------------------------------------------------------
__global__ void cvt_kernel(const float* in, half_t* out, long long n) {
  long long i = (long long)blockIdx.x * 256 + threadIdx.x;
  if (i < n) out[i] = (half_t)in[i];
}
__global__ void add_kernel(const float* a, const float* b, float* o, long long n) {
  long long i = (long long)blockIdx.x * 256 + threadIdx.x;
  if (i < n) o[i] = a[i] + b[i];
}
__global__ void zero_kernel(float* p, long long n) {
  long long i = (long long)blockIdx.x * 256 + threadIdx.x;
  if (i < n) p[i] = 0.f;
}
__global__ void bcast_q_kernel(const float* q, float* out, long long n) {
  long long i = (long long)blockIdx.x * 256 + threadIdx.x;
  if (i < n) out[i] = q[i % (100 * 256)];
}
__global__ void mem_pos_kernel(const float* memf, const float* pos, float* of, half_t* oh) {
  long long i = (long long)blockIdx.x * 256 + threadIdx.x;
  const long long N = 8LL * 1024 * 256;
  if (i >= N) return;
  int c = (int)(i % 256);
  long long bt = i / 256;
  int t = (int)(bt % 1024), b = (int)(bt / 1024);
  float v = memf[((long long)b * 256 + c) * 1024 + t] + pos[(long long)t * 256 + c];
  of[i] = v; oh[i] = (half_t)v;
}
__global__ void copy_chan_kernel(const float* src, float* dst, int B_, int Cs, int HW, int Cd, int coff) {
  long long i = (long long)blockIdx.x * 256 + threadIdx.x;
  long long N = (long long)B_ * Cs * HW;
  if (i >= N) return;
  int hw = (int)(i % HW);
  long long t = i / HW;
  int c = (int)(t % Cs), b = (int)(t / Cs);
  dst[(((long long)b * Cd + coff + c) * HW) + hw] = src[i];
}

__global__ void layernorm_kernel(const float* x, const float* g, const float* b, half_t* out) {
  int row = blockIdx.x, c = threadIdx.x;           // 256 threads, C=256
  const float* r = x + (long long)row * 256;
  __shared__ float red[256];
  float v = r[c];
  red[c] = v; __syncthreads();
  for (int s = 128; s > 0; s >>= 1) { if (c < s) red[c] += red[c + s]; __syncthreads(); }
  float mu = red[0] / 256.f; __syncthreads();
  float d = v - mu;
  red[c] = d * d; __syncthreads();
  for (int s = 128; s > 0; s >>= 1) { if (c < s) red[c] += red[c + s]; __syncthreads(); }
  float inv = rsqrtf(red[0] / 256.f + 1e-5f);
  out[(long long)row * 256 + c] = (half_t)(d * inv * g[c] + b[c]);
}

__global__ void softmax_rows_kernel(const float* in, half_t* out, int n) {
  int row = blockIdx.x;
  const float* r = in + (long long)row * n;
  __shared__ float red[256];
  float mx = -1e30f;
  for (int i = threadIdx.x; i < n; i += 256) mx = fmaxf(mx, r[i]);
  red[threadIdx.x] = mx; __syncthreads();
  for (int s = 128; s > 0; s >>= 1) { if (threadIdx.x < s) red[threadIdx.x] = fmaxf(red[threadIdx.x], red[threadIdx.x + s]); __syncthreads(); }
  mx = red[0]; __syncthreads();
  float sum = 0.f;
  for (int i = threadIdx.x; i < n; i += 256) sum += __expf(r[i] - mx);
  red[threadIdx.x] = sum; __syncthreads();
  for (int s = 128; s > 0; s >>= 1) { if (threadIdx.x < s) red[threadIdx.x] += red[threadIdx.x + s]; __syncthreads(); }
  float inv = 1.f / red[0];
  half_t* o = out + (long long)row * n;
  for (int i = threadIdx.x; i < n; i += 256) o[i] = (half_t)(__expf(r[i] - mx) * inv);
}

__global__ void depth_residual_kernel(const float* hist, const float* rq, const float* ks,
                                      float* out, int d, int rows) {
  int row = blockIdx.x, c = threadIdx.x;   // 256 threads, C=256
  __shared__ float red[256]; __shared__ float lg[8];
  float hv[8];
  for (int j = 0; j < d; ++j) {
    float h = hist[((long long)j * rows + row) * 256 + c];
    hv[j] = h;
    red[c] = h * h; __syncthreads();
    for (int s = 128; s > 0; s >>= 1) { if (c < s) red[c] += red[c + s]; __syncthreads(); }
    float rms = rsqrtf(red[0] / 256.f + 1e-6f); __syncthreads();
    red[c] = rq[c] * h * rms * ks[c]; __syncthreads();
    for (int s = 128; s > 0; s >>= 1) { if (c < s) red[c] += red[c + s]; __syncthreads(); }
    if (c == 0) lg[j] = red[0];
    __syncthreads();
  }
  long long o = (long long)row * 256 + c;
  if (d == 1) { out[o] = hv[0]; return; }
  float mx = -1e30f; for (int j = 0; j < d; ++j) mx = fmaxf(mx, lg[j]);
  float s = 0.f;     for (int j = 0; j < d; ++j) s += __expf(lg[j] - mx);
  float acc = 0.f;   for (int j = 0; j < d; ++j) acc += __expf(lg[j] - mx) / s * hv[j];
  out[o] = acc;
}

// TTT analytic inner gradient: loss = sum_b mean_q (1 - sigmoid(mlp(q)))
__global__ void ttt_ds_kernel(const float* s, float* ds, int n) {
  int i = blockIdx.x * 256 + threadIdx.x;
  if (i >= n) return;
  float p = 1.f / (1.f + __expf(-s[i]));
  ds[i] = -p * (1.f - p) / 100.f;
}
__global__ void ttt_dh_kernel(const half_t* h, const float* ds, const float* w2,
                              half_t* dpre, long long n) {
  long long i = (long long)blockIdx.x * 256 + threadIdx.x;
  if (i >= n) return;
  int row = (int)(i >> 8), c = (int)(i & 255);
  float hv = (float)h[i];
  dpre[i] = (half_t)((hv > 0.f) ? ds[row] * w2[c] : 0.f);
}
__global__ void ttt_update_kernel(float* q, float* v, const float* dq,
                                  const float* raw_lr, const float* raw_mom, long long n) {
  long long i = (long long)blockIdx.x * 256 + threadIdx.x;
  if (i >= n) return;
  float lr  = log1pf(__expf(raw_lr[0])) + 1e-8f;           // softplus
  float mom = 1.f / (1.f + __expf(-raw_mom[0])) + 1e-8f;   // sigmoid
  float nv = mom * v[i] - lr * dq[i];
  v[i] = nv; q[i] += nv;
}

// ---------------------------------------------------------------------------
// Backbone convs (eval BN + ReLU fused), 3x3 pad1 / ConvTranspose k4 s2 p1
// ---------------------------------------------------------------------------
__global__ void conv3_bn_relu_kernel(const float* in, const float* w, const float* cb,
                                     const float* g, const float* be, const float* m, const float* vv,
                                     float* out, int B_, int Ci, int Co, int Hi, int Wi,
                                     int st, int Ho, int Wo) {
  long long i = (long long)blockIdx.x * 256 + threadIdx.x;
  long long N = (long long)B_ * Co * Ho * Wo;
  if (i >= N) return;
  int x = (int)(i % Wo); long long t = i / Wo;
  int y = (int)(t % Ho); t /= Ho;
  int co = (int)(t % Co); int b = (int)(t / Co);
  float acc = cb[co];
  for (int ci = 0; ci < Ci; ++ci)
    for (int ky = 0; ky < 3; ++ky) {
      int iy = y * st + ky - 1;
      if (iy < 0 || iy >= Hi) continue;
      for (int kx = 0; kx < 3; ++kx) {
        int ix = x * st + kx - 1;
        if (ix < 0 || ix >= Wi) continue;
        acc += in[(((long long)b * Ci + ci) * Hi + iy) * Wi + ix] *
               w[(((long long)co * Ci + ci) * 3 + ky) * 3 + kx];
      }
    }
  float bn = (acc - m[co]) * rsqrtf(vv[co] + 1e-5f) * g[co] + be[co];
  out[i] = fmaxf(bn, 0.f);
}

__global__ void convt_bn_relu_kernel(const float* in, const float* w, const float* cb,
                                     const float* g, const float* be, const float* m, const float* vv,
                                     float* out, int B_, int Ci, int Co, int Hi, int Wi,
                                     int Ho, int Wo) {
  long long i = (long long)blockIdx.x * 256 + threadIdx.x;
  long long N = (long long)B_ * Co * Ho * Wo;
  if (i >= N) return;
  int x = (int)(i % Wo); long long t = i / Wo;
  int y = (int)(t % Ho); t /= Ho;
  int co = (int)(t % Co); int b = (int)(t / Co);
  float acc = cb[co];
  for (int ci = 0; ci < Ci; ++ci)
    for (int ky = 0; ky < 4; ++ky) {
      int ty = y + 1 - ky;
      if (ty < 0 || (ty & 1)) continue;
      int iy = ty >> 1;
      if (iy >= Hi) continue;
      for (int kx = 0; kx < 4; ++kx) {
        int tx = x + 1 - kx;
        if (tx < 0 || (tx & 1)) continue;
        int ix = tx >> 1;
        if (ix >= Wi) continue;
        acc += in[(((long long)b * Ci + ci) * Hi + iy) * Wi + ix] *
               w[(((long long)ci * Co + co) * 4 + ky) * 4 + kx];   // torch (Cin,Cout,4,4)
      }
    }
  float bn = (acc - m[co]) * rsqrtf(vv[co] + 1e-5f) * g[co] + be[co];
  out[i] = fmaxf(bn, 0.f);
}

// ---------------------------------------------------------------------------
// Orchestration
// ---------------------------------------------------------------------------
extern "C" void kernel_launch(void* const* d_in, const int* in_sizes, int n_in,
                              void* d_out, int out_size, void* d_ws, size_t ws_size,
                              hipStream_t stream) {
  (void)in_sizes; (void)n_in; (void)out_size; (void)ws_size;
  const float* X = (const float*)d_in[0];
  auto F = [&](int i) -> const float* { return (const float*)d_in[i]; };

  // Leaf order: x; bb (7 groups x {w,cb,bn.g,bn.b,bn.m,bn.v}); 6 layers x 20;
  // queries, pos, res_q, res_k, seed{w1,b1,w2,b2}, mask_head{w1,b1,w2,b2}.
  auto BBW = [&](int g){ return F(1 + g*6 + 0); };
  auto BBC = [&](int g){ return F(1 + g*6 + 1); };
  auto BBG = [&](int g){ return F(1 + g*6 + 2); };
  auto BBB = [&](int g){ return F(1 + g*6 + 3); };
  auto BBM = [&](int g){ return F(1 + g*6 + 4); };
  auto BBV = [&](int g){ return F(1 + g*6 + 5); };
  auto LY  = [&](int l, int o){ return F(43 + l*20 + o); };
  const float* Pq   = F(163); const float* Ppos = F(164);
  const float* Prq  = F(165); const float* Prk  = F(166);
  const float* Sw1  = F(167); const float* Sb1  = F(168);
  const float* Sw2  = F(169); const float* Sb2  = F(170);
  const float* Mw1  = F(171); const float* Mb1  = F(172);
  const float* Mw2  = F(173); const float* Mb2  = F(174);

  size_t off = 0;
  auto A_ = [&](long long elems, int esz) -> void* {
    void* p = (char*)d_ws + off;
    off += ((size_t)elems * esz + 255) & ~(size_t)255;
    return p;
  };
  float*  stem  = (float*)A_(8LL*32*128*128, 4);
  float*  mid   = (float*)A_(8LL*64*64*64, 4);
  float*  memf  = (float*)A_(8LL*256*32*32, 4);
  float*  u1    = (float*)A_(8LL*64*64*64, 4);
  float*  cat1  = (float*)A_(8LL*128*64*64, 4);
  float*  f1b   = (float*)A_(8LL*64*64*64, 4);
  float*  u2    = (float*)A_(8LL*32*128*128, 4);
  float*  cat2  = (float*)A_(8LL*64*128*128, 4);
  float*  maskf = (float*)A_(8LL*256*128*128, 4);
  half_t* maskh = (half_t*)A_(8LL*256*16384, 2);
  float*  memory= (float*)A_(8LL*1024*256, 4);
  half_t* memh  = (half_t*)A_(8LL*1024*256, 2);
  float*  hist  = (float*)A_(7LL*800*256, 4);
  float*  inp   = (float*)A_(800LL*256, 4);
  float*  tgt   = (float*)A_(800LL*256, 4);
  float*  proj  = (float*)A_(800LL*256, 4);
  half_t* th    = (half_t*)A_(800LL*256, 2);
  half_t* qh    = (half_t*)A_(800LL*256, 2);
  half_t* kh    = (half_t*)A_(800LL*256, 2);
  half_t* vhb   = (half_t*)A_(800LL*256, 2);
  half_t* memk  = (half_t*)A_(8192LL*256, 2);
  half_t* memv  = (half_t*)A_(8192LL*256, 2);
  float*  scores= (float*)A_(8LL*8*100*1024, 4);
  half_t* att   = (half_t*)A_(8LL*8*100*1024, 2);
  half_t* attout= (half_t*)A_(800LL*256, 2);
  half_t* ffh   = (half_t*)A_(800LL*1024, 2);
  half_t* hseed = (half_t*)A_(800LL*256, 2);
  float*  sbuf  = (float*)A_(800, 4);
  float*  dsb   = (float*)A_(800, 4);
  half_t* dpre  = (half_t*)A_(800LL*256, 2);
  float*  dqb   = (float*)A_(800LL*256, 4);
  float*  vel   = (float*)A_(800LL*256, 4);
  half_t* tgth  = (half_t*)A_(800LL*256, 2);
  half_t* allh  = (half_t*)A_(6LL*800*256, 2);
  half_t* mh    = (half_t*)A_(4800LL*256, 2);
  half_t* emb   = (half_t*)A_(4800LL*256, 2);
  half_t *saW[6], *saO[6], *caW[6], *caO[6], *l1W[6], *l2W[6];
  for (int l = 0; l < 6; ++l) {
    saW[l] = (half_t*)A_(768LL*256, 2);  saO[l] = (half_t*)A_(256LL*256, 2);
    caW[l] = (half_t*)A_(768LL*256, 2);  caO[l] = (half_t*)A_(256LL*256, 2);
    l1W[l] = (half_t*)A_(1024LL*256, 2); l2W[l] = (half_t*)A_(256LL*1024, 2);
  }
  half_t* SW1h = (half_t*)A_(256LL*256, 2);
  half_t* SW2h = (half_t*)A_(256, 2);
  half_t* MW1h = (half_t*)A_(256LL*256, 2);
  half_t* MW2h = (half_t*)A_(256LL*256, 2);

  auto EL = [&](long long n) { return dim3((unsigned)((n + 255) / 256)); };
  auto cvt = [&](const float* a, half_t* b, long long n) {
    cvt_kernel<<<EL(n), 256, 0, stream>>>(a, b, n);
  };
  auto gemm = [&](const half_t* A, const half_t* Bp, const float* bias,
                  float* Cf, half_t* Ch, int M, int N, int K,
                  int lda, int ldb, int ldc,
                  long long sAo, long long sAi, long long sBo, long long sBi,
                  long long sCo, long long sCi, int inner, int nb,
                  int btrans, int act, float alpha) {
    GemmP p{A, Bp, bias, Cf, Ch, M, N, K, lda, ldb, ldc,
            sAo, sAi, sBo, sBi, sCo, sCi, inner, btrans, act, alpha};
    dim3 g((unsigned)((N + 63) / 64), (unsigned)((M + 127) / 128), (unsigned)nb);
    wmma_gemm_kernel<<<g, 256, 0, stream>>>(p);
  };

  // --- weight f32 -> f16 ---
  for (int l = 0; l < 6; ++l) {
    cvt(LY(l, 2),  saW[l], 768LL*256);  cvt(LY(l, 4),  saO[l], 256LL*256);
    cvt(LY(l, 8),  caW[l], 768LL*256);  cvt(LY(l, 10), caO[l], 256LL*256);
    cvt(LY(l, 14), l1W[l], 1024LL*256); cvt(LY(l, 16), l2W[l], 256LL*1024);
  }
  cvt(Sw1, SW1h, 256LL*256); cvt(Sw2, SW2h, 256);
  cvt(Mw1, MW1h, 256LL*256); cvt(Mw2, MW2h, 256LL*256);

  // --- backbone ---
  conv3_bn_relu_kernel<<<EL(8LL*32*128*128), 256, 0, stream>>>(X, BBW(0), BBC(0), BBG(0), BBB(0), BBM(0), BBV(0), stem, 8, 3, 32, 128, 128, 1, 128, 128);
  conv3_bn_relu_kernel<<<EL(8LL*64*64*64), 256, 0, stream>>>(stem, BBW(1), BBC(1), BBG(1), BBB(1), BBM(1), BBV(1), mid, 8, 32, 64, 128, 128, 2, 64, 64);
  conv3_bn_relu_kernel<<<EL(8LL*256*32*32), 256, 0, stream>>>(mid, BBW(2), BBC(2), BBG(2), BBB(2), BBM(2), BBV(2), memf, 8, 64, 256, 64, 64, 2, 32, 32);
  convt_bn_relu_kernel<<<EL(8LL*64*64*64), 256, 0, stream>>>(memf, BBW(3), BBC(3), BBG(3), BBB(3), BBM(3), BBV(3), u1, 8, 256, 64, 32, 32, 64, 64);
  copy_chan_kernel<<<EL(8LL*64*4096), 256, 0, stream>>>(u1, cat1, 8, 64, 4096, 128, 0);
  copy_chan_kernel<<<EL(8LL*64*4096), 256, 0, stream>>>(mid, cat1, 8, 64, 4096, 128, 64);
  conv3_bn_relu_kernel<<<EL(8LL*64*64*64), 256, 0, stream>>>(cat1, BBW(4), BBC(4), BBG(4), BBB(4), BBM(4), BBV(4), f1b, 8, 128, 64, 64, 64, 1, 64, 64);
  convt_bn_relu_kernel<<<EL(8LL*32*128*128), 256, 0, stream>>>(f1b, BBW(5), BBC(5), BBG(5), BBB(5), BBM(5), BBV(5), u2, 8, 64, 32, 64, 64, 128, 128);
  copy_chan_kernel<<<EL(8LL*32*16384), 256, 0, stream>>>(u2, cat2, 8, 32, 16384, 64, 0);
  copy_chan_kernel<<<EL(8LL*32*16384), 256, 0, stream>>>(stem, cat2, 8, 32, 16384, 64, 32);
  conv3_bn_relu_kernel<<<EL(8LL*256*16384), 256, 0, stream>>>(cat2, BBW(6), BBC(6), BBG(6), BBB(6), BBM(6), BBV(6), maskf, 8, 64, 256, 128, 128, 1, 128, 128);
  cvt(maskf, maskh, 8LL*256*16384);

  mem_pos_kernel<<<EL(8LL*1024*256), 256, 0, stream>>>(memf, Ppos, memory, memh);
  bcast_q_kernel<<<EL(800LL*256), 256, 0, stream>>>(Pq, hist, 800LL*256);   // history[0]

  const long long NR = 800LL * 256;
  const float scl = 0.17677669529663687f;   // 1/sqrt(32)

  for (int l = 0; l < 6; ++l) {
    // depth residual over history[0..l]
    depth_residual_kernel<<<800, 256, 0, stream>>>(hist, Prq + l * 256, Prk, inp, l + 1, 800);

    // --- self attention ---
    layernorm_kernel<<<800, 256, 0, stream>>>(inp, LY(l, 0), LY(l, 1), th);
    gemm(th, saW[l],             LY(l, 3),       nullptr, qh, 800, 256, 256, 256, 256, 256, 0,0,0,0,0,0, 1, 1, 1, 0, 1.f);
    gemm(th, saW[l] + 256*256,   LY(l, 3) + 256, nullptr, kh, 800, 256, 256, 256, 256, 256, 0,0,0,0,0,0, 1, 1, 1, 0, 1.f);
    gemm(th, saW[l] + 512*256,   LY(l, 3) + 512, nullptr, vhb,800, 256, 256, 256, 256, 256, 0,0,0,0,0,0, 1, 1, 1, 0, 1.f);
    gemm(qh, kh, nullptr, scores, nullptr, 100, 100, 32, 256, 256, 100,
         25600, 32, 25600, 32, 80000, 10000, 8, 64, 1, 0, scl);
    softmax_rows_kernel<<<6400, 256, 0, stream>>>(scores, att, 100);
    gemm(att, vhb, nullptr, nullptr, attout, 100, 32, 100, 100, 256, 256,
         80000, 10000, 25600, 32, 25600, 32, 8, 64, 0, 0, 1.f);
    gemm(attout, saO[l], LY(l, 5), proj, nullptr, 800, 256, 256, 256, 256, 256, 0,0,0,0,0,0, 1, 1, 1, 0, 1.f);
    add_kernel<<<EL(NR), 256, 0, stream>>>(inp, proj, tgt, NR);

    // --- cross attention ---
    layernorm_kernel<<<800, 256, 0, stream>>>(tgt, LY(l, 6), LY(l, 7), th);
    gemm(th,   caW[l],           LY(l, 9),       nullptr, qh,   800,  256, 256, 256, 256, 256, 0,0,0,0,0,0, 1, 1, 1, 0, 1.f);
    gemm(memh, caW[l] + 256*256, LY(l, 9) + 256, nullptr, memk, 8192, 256, 256, 256, 256, 256, 0,0,0,0,0,0, 1, 1, 1, 0, 1.f);
    gemm(memh, caW[l] + 512*256, LY(l, 9) + 512, nullptr, memv, 8192, 256, 256, 256, 256, 256, 0,0,0,0,0,0, 1, 1, 1, 0, 1.f);
    gemm(qh, memk, nullptr, scores, nullptr, 100, 1024, 32, 256, 256, 1024,
         25600, 32, 262144, 32, 819200, 102400, 8, 64, 1, 0, scl);
    softmax_rows_kernel<<<6400, 256, 0, stream>>>(scores, att, 1024);
    gemm(att, memv, nullptr, nullptr, attout, 100, 32, 1024, 1024, 256, 256,
         819200, 102400, 262144, 32, 25600, 32, 8, 64, 0, 0, 1.f);
    gemm(attout, caO[l], LY(l, 11), proj, nullptr, 800, 256, 256, 256, 256, 256, 0,0,0,0,0,0, 1, 1, 1, 0, 1.f);
    add_kernel<<<EL(NR), 256, 0, stream>>>(tgt, proj, tgt, NR);

    // --- TTT inner loop (3 steps of momentum SGD on queries) ---
    zero_kernel<<<EL(NR), 256, 0, stream>>>(vel, NR);
    for (int s = 0; s < 3; ++s) {
      cvt(tgt, tgth, NR);
      gemm(tgth, SW1h, Sb1, nullptr, hseed, 800, 256, 256, 256, 256, 256, 0,0,0,0,0,0, 1, 1, 1, 1, 1.f);
      gemm(hseed, SW2h, Sb2, sbuf, nullptr, 800, 1, 256, 256, 256, 1, 0,0,0,0,0,0, 1, 1, 1, 0, 1.f);
      ttt_ds_kernel<<<EL(800), 256, 0, stream>>>(sbuf, dsb, 800);
      ttt_dh_kernel<<<EL(NR), 256, 0, stream>>>(hseed, dsb, Sw2, dpre, NR);
      gemm(dpre, SW1h, nullptr, dqb, nullptr, 800, 256, 256, 256, 256, 256, 0,0,0,0,0,0, 1, 1, 0, 0, 1.f);
      ttt_update_kernel<<<EL(NR), 256, 0, stream>>>(tgt, vel, dqb, LY(l, 18), LY(l, 19), NR);
    }

    // --- FFN ---
    layernorm_kernel<<<800, 256, 0, stream>>>(tgt, LY(l, 12), LY(l, 13), th);
    gemm(th,  l1W[l], LY(l, 15), nullptr, ffh, 800, 1024, 256, 256, 256, 1024, 0,0,0,0,0,0, 1, 1, 1, 2, 1.f);
    gemm(ffh, l2W[l], LY(l, 17), proj, nullptr, 800, 256, 1024, 1024, 1024, 256, 0,0,0,0,0,0, 1, 1, 1, 0, 1.f);
    add_kernel<<<EL(NR), 256, 0, stream>>>(tgt, proj, hist + (long long)(l + 1) * NR, NR);
  }

  // --- mask head on stacked layer outputs [6,8,100,256] ---
  cvt(hist + NR, allh, 6 * NR);
  gemm(allh, MW1h, Mb1, nullptr, mh,  4800, 256, 256, 256, 256, 256, 0,0,0,0,0,0, 1, 1, 1, 1, 1.f);
  gemm(mh,   MW2h, Mb2, nullptr, emb, 4800, 256, 256, 256, 256, 256, 0,0,0,0,0,0, 1, 1, 1, 0, 1.f);

  // --- final einsum lbqc,bchw->lbqhw : 48 GEMMs [100x256]*[256x16384] ---
  gemm(emb, maskh, nullptr, (float*)d_out, nullptr, 100, 16384, 256,
       256, 16384, 16384,
       204800, 25600,          // A: per-layer, per-batch strides
       0, 4194304,             // B: shared across layers, per-batch stride
       13107200, 1638400,      // C: [l][b][q][hw]
       8, 48, 0, 0, 1.f);
}
